// LSTMMemOptDecoder_24678882083433
// MI455X (gfx1250) — matrix operand order repaired and checked by reading, not compiled, not run on hardware
//
#include <hip/hip_runtime.h>
#include <hip/hip_bf16.h>
#include <math.h>
#include <cstddef>

// ---------------------------------------------------------------------------
// Problem constants (from reference): TX=160, TY=40, B=16, H=256, C=512,
// IN=128, V=30000.
// ---------------------------------------------------------------------------
#define TXX 160
#define TYY 40
#define BB  16
#define HH  256
#define CC  512
#define ININ 128
#define VV  30000

typedef __bf16 bf16_t;
typedef bf16_t v16bf __attribute__((ext_vector_type(16)));
typedef bf16_t v8bf  __attribute__((ext_vector_type(8)));
typedef float  v8f   __attribute__((ext_vector_type(8)));

__device__ __forceinline__ float sigm(float x) { return 1.0f / (1.0f + __expf(-x)); }

// ---------------------------------------------------------------------------
// WMMA GEMM with N-tile blocking:
//   D[M x N] = A[M x K](bf16, row-major) @ W[N x K]^T(bf16, row-major) + bias.
// One wave computes NT consecutive 16x16 tiles along N, reusing each A
// fragment NT times (load pressure (2+2*NT)/NT b128 per wmma instead of 4).
// Requirements: M,N multiples of 16; (N/16) % NT == 0; K multiple of 32;
// all row strides 16-byte aligned.
//
// A-fragment layout (ISA 7.12.2, 16-bit A 16x32): lane L holds row M=L%16;
//   elems 0..7  -> K = (L/16)*8 + 0..7
//   elems 8..15 -> K = (L/16)*8 + 16..23
// B-fragment layout (32x16): lane L holds col N=L%16;
//   elems 0..7  -> K = (L/16)*8 + 0..7
//   elems 8..15 -> K = (L/16)*8 + 16..23   (same as A)
// C/D layout: lane L holds col N=L%16; vgpr r holds row r + 8*(L/16).
// ---------------------------------------------------------------------------
template <int NT>
__global__ void wmma_gemm_bias(const bf16_t* __restrict__ A, int lda,
                               const bf16_t* __restrict__ W, int ldw,
                               const float* __restrict__ bias,
                               float* __restrict__ D, int ldd,
                               int M, int N, int K)
{
    const int wave = (int)((blockIdx.x * blockDim.x + threadIdx.x) >> 5);
    const int lane = (int)(threadIdx.x & 31);
    const int groups_n = (N >> 4) / NT;
    const int tiles_m  = (M + 15) >> 4;
    if (wave >= tiles_m * groups_n) return;
    const int tm  = wave / groups_n;
    const int tn0 = (wave - tm * groups_n) * NT;

    const int lhalf = lane >> 4;   // 0 or 1
    const int lmod  = lane & 15;

    const bf16_t* ap = A + (size_t)(tm * 16 + lmod) * lda + (lhalf << 3);
    const bf16_t* wp = W + (size_t)(tn0 * 16 + lmod) * ldw + (lhalf << 3);
    const size_t  wstep = (size_t)16 * ldw;   // next N tile

    v8f acc[NT] = {};
    for (int k = 0; k < K; k += 32) {
        v8bf alo = *(const v8bf*)(ap + k);
        v8bf ahi = *(const v8bf*)(ap + k + 16);
        v16bf af = __builtin_shufflevector(alo, ahi, 0,1,2,3,4,5,6,7,
                                                     8,9,10,11,12,13,14,15);
        // keep the weight stream ahead (emits global_prefetch_b8 on gfx1250)
        __builtin_prefetch(wp + k + 512, 0, 0);
        #pragma unroll
        for (int i = 0; i < NT; ++i) {
            const bf16_t* wpi = wp + (size_t)i * wstep;
            v8bf blo = *(const v8bf*)(wpi + k);
            v8bf bhi = *(const v8bf*)(wpi + k + 16);
            v16bf bf_ = __builtin_shufflevector(blo, bhi, 0,1,2,3,4,5,6,7,
                                                          8,9,10,11,12,13,14,15);
            acc[i] = __builtin_amdgcn_wmma_f32_16x16x32_bf16(
                /*neg_a=*/false, af, /*neg_b=*/false, bf_,
                /*c_mod=*/(short)0, acc[i], /*reuse_a=*/false, /*reuse_b=*/false);
        }
    }

    #pragma unroll
    for (int i = 0; i < NT; ++i) {
        const int col = (tn0 + i) * 16 + lmod;
        const float bv = bias ? bias[col] : 0.0f;
        #pragma unroll
        for (int r = 0; r < 8; ++r) {
            int row = tm * 16 + r + (lhalf << 3);
            if (row < M) D[(size_t)row * ldd + col] = acc[i][r] + bv;
        }
    }
}

// ---------------------------------------------------------------------------
// One-time conversion / packing kernels.
// ---------------------------------------------------------------------------
__global__ void cvt_bf16(const float* __restrict__ s, bf16_t* __restrict__ d, int n)
{
    int i = blockIdx.x * blockDim.x + threadIdx.x;
    if (i < n) d[i] = (bf16_t)s[i];
}

// dst row r (kt = ka+kb wide): [A row r (ka) | B row r (kb)]   (fp32 -> bf16)
__global__ void pack2_bf16(const float* __restrict__ A, int ka,
                           const float* __restrict__ B, int kb,
                           bf16_t* __restrict__ d, int rows)
{
    int i = blockIdx.x * blockDim.x + threadIdx.x;
    int kt = ka + kb;
    if (i >= rows * kt) return;
    int r = i / kt, k = i - r * kt;
    float v = (k < ka) ? A[(size_t)r * ka + k] : B[(size_t)r * kb + (k - ka)];
    d[i] = (bf16_t)v;
}

__global__ void init_state(const float* __restrict__ h0, const float* __restrict__ c0,
                           const float* __restrict__ cov0,
                           const float* __restrict__ b_ih, const float* __restrict__ b_hh,
                           float* __restrict__ h, float* __restrict__ c,
                           float* __restrict__ acc, float* __restrict__ blstm,
                           float* __restrict__ covacc)
{
    int i = blockIdx.x * blockDim.x + threadIdx.x;
    if (i < BB * HH) { h[i] = h0[i]; c[i] = c0[i]; }
    if (i < BB * TXX) acc[i] = cov0[i];
    if (i < 4 * HH) blstm[i] = b_ih[i] + b_hh[i];
    if (i == 0) *covacc = 0.0f;
}

// ---------------------------------------------------------------------------
// Per-step kernels.
// ---------------------------------------------------------------------------
// Build xh = [y_t | h] in bf16, and fill the y_t slice of feats.
__global__ void prep_step(const bf16_t* __restrict__ yemb_bf, const float* __restrict__ h,
                          bf16_t* __restrict__ xh, bf16_t* __restrict__ feats, int t)
{
    int i = blockIdx.x * blockDim.x + threadIdx.x;  // BB * (IN+H) = 6144
    if (i >= BB * (ININ + HH)) return;
    int b = i / (ININ + HH), k = i - b * (ININ + HH);
    bf16_t v;
    if (k < ININ) {
        v = yemb_bf[((size_t)t * BB + b) * ININ + k];
        feats[(size_t)b * (HH + CC + ININ) + HH + CC + k] = v;
    } else {
        v = (bf16_t)h[b * HH + (k - ININ)];
    }
    xh[(size_t)b * (ININ + HH) + k] = v;
}

// First LSTM cell: gates order i,f,g,o.  Writes h1/c1 (fp32), s2=[h1|c1] bf16,
// and the h1 slice of ha=[h1|atted].
__global__ void lstm_cell1(const float* __restrict__ gates, const float* __restrict__ h,
                           const float* __restrict__ c, const float* __restrict__ ymask,
                           float* __restrict__ h1, float* __restrict__ c1,
                           bf16_t* __restrict__ s2, bf16_t* __restrict__ ha, int t)
{
    int i = blockIdx.x * blockDim.x + threadIdx.x;
    if (i >= BB * HH) return;
    int b = i >> 8, j = i & (HH - 1);
    const float* g = gates + (size_t)b * 4 * HH;
    float gi = g[j], gf = g[HH + j], gg = g[2 * HH + j], go = g[3 * HH + j];
    float cn = sigm(gf) * c[i] + sigm(gi) * tanhf(gg);
    float hn = sigm(go) * tanhf(cn);
    float ym = ymask[t * BB + b];
    hn = ym * hn + (1.0f - ym) * h[i];
    cn = ym * cn + (1.0f - ym) * c[i];
    h1[i] = hn; c1[i] = cn;
    s2[(size_t)b * (2 * HH) + j]      = (bf16_t)hn;
    s2[(size_t)b * (2 * HH) + HH + j] = (bf16_t)cn;
    ha[(size_t)b * (HH + CC) + j]     = (bf16_t)hn;
}

// Attention scores: one wave per (tx,b); sc = u . (tanh(pctx+hq+cov*wcov)) * xm
__global__ void attn_score(const float* __restrict__ pctx, const float* __restrict__ hq,
                           const float* __restrict__ acc, const float* __restrict__ wcov,
                           const float* __restrict__ uvec, const float* __restrict__ xmask,
                           float* __restrict__ sc)
{
    int wave = (int)((blockIdx.x * blockDim.x + threadIdx.x) >> 5);
    int lane = (int)(threadIdx.x & 31);
    if (wave >= TXX * BB) return;
    int b = wave & (BB - 1);
    int tx = wave >> 4;
    const float* pr = pctx + (size_t)wave * CC;
    const float* hr = hq + (size_t)b * CC;
    float cv = acc[b * TXX + tx];
    float s = 0.0f;
    for (int cc = lane; cc < CC; cc += 32)
        s += tanhf(pr[cc] + hr[cc] + cv * wcov[cc]) * uvec[cc];
    for (int o = 16; o; o >>= 1) s += __shfl_down(s, o, 32);
    if (lane == 0) sc[wave] = s * xmask[wave];
}

// Softmax over TX per batch, context-weighted sum, coverage update + cost.
__global__ void attn_soft(const float* __restrict__ sc, const float* __restrict__ xmask,
                          const float* __restrict__ context, float* __restrict__ acc,
                          bf16_t* __restrict__ ha, bf16_t* __restrict__ feats,
                          float* __restrict__ covacc)
{
    __shared__ float s_w[TXX];
    __shared__ float red[256];
    int b = blockIdx.x, tid = threadIdx.x;

    float v = -1e30f;
    if (tid < TXX) { v = sc[tid * BB + b]; s_w[tid] = v; }
    red[tid] = v; __syncthreads();
    for (int o = 128; o; o >>= 1) { if (tid < o) red[tid] = fmaxf(red[tid], red[tid + o]); __syncthreads(); }
    float mx = red[0]; __syncthreads();

    float e = 0.0f;
    if (tid < TXX) { e = __expf(s_w[tid] - mx) * xmask[tid * BB + b]; s_w[tid] = e; }
    red[tid] = e; __syncthreads();
    for (int o = 128; o; o >>= 1) { if (tid < o) red[tid] += red[tid + o]; __syncthreads(); }
    float inv = 1.0f / red[0]; __syncthreads();

    float covc = 0.0f;
    if (tid < TXX) {
        float w = s_w[tid] * inv;
        s_w[tid] = w;
        float a0 = acc[b * TXX + tid];
        covc = fminf(w, a0);
        acc[b * TXX + tid] = a0 + w;
    }
    red[tid] = covc; __syncthreads();
    for (int o = 128; o; o >>= 1) { if (tid < o) red[tid] += red[tid + o]; __syncthreads(); }
    if (tid == 0) atomicAdd(covacc, red[0]);
    __syncthreads();

    for (int cidx = tid; cidx < CC; cidx += 256) {
        float s = 0.0f;
        for (int tx = 0; tx < TXX; ++tx)
            s += s_w[tx] * context[((size_t)tx * BB + b) * CC + cidx];
        bf16_t bv = (bf16_t)s;
        ha[(size_t)b * (HH + CC) + HH + cidx]           = bv;
        feats[(size_t)b * (HH + CC + ININ) + HH + cidx] = bv;
    }
}

// Second LSTM cell: ifoc order i,f,o,g (per reference!).
__global__ void lstm_cell2(const float* __restrict__ ifoc, const float* __restrict__ h1,
                           const float* __restrict__ c1, const float* __restrict__ ymask,
                           float* __restrict__ h, float* __restrict__ c,
                           bf16_t* __restrict__ feats, int t)
{
    int i = blockIdx.x * blockDim.x + threadIdx.x;
    if (i >= BB * HH) return;
    int b = i >> 8, j = i & (HH - 1);
    const float* g = ifoc + (size_t)b * 4 * HH;
    float i2 = g[j], f2 = g[HH + j], o2 = g[2 * HH + j], g2 = g[3 * HH + j];
    float cn = sigm(f2) * c1[i] + sigm(i2) * tanhf(g2);
    float hn = sigm(o2) * tanhf(cn);
    float ym = ymask[t * BB + b];
    cn = ym * cn + (1.0f - ym) * c1[i];
    hn = ym * hn + (1.0f - ym) * h1[i];
    h[i] = hn; c[i] = cn;
    feats[(size_t)b * (HH + CC + ININ) + j] = (bf16_t)hn;
}

// NLL cost via log-softmax over V, one block per batch element.
__global__ void nll_cost(const float* __restrict__ logits, const int* __restrict__ yidx,
                         float* __restrict__ costs, int t)
{
    __shared__ float red[256];
    int b = blockIdx.x, tid = threadIdx.x;
    const float* row = logits + (size_t)b * VV;
    float mx = -1e30f;
    for (int v = tid; v < VV; v += 256) mx = fmaxf(mx, row[v]);
    red[tid] = mx; __syncthreads();
    for (int o = 128; o; o >>= 1) { if (tid < o) red[tid] = fmaxf(red[tid], red[tid + o]); __syncthreads(); }
    mx = red[0]; __syncthreads();
    float s = 0.0f;
    for (int v = tid; v < VV; v += 256) s += __expf(row[v] - mx);
    red[tid] = s; __syncthreads();
    for (int o = 128; o; o >>= 1) { if (tid < o) red[tid] += red[tid + o]; __syncthreads(); }
    if (tid == 0) {
        float lse = __logf(red[0]) + mx;
        int yi = yidx[t * BB + b];
        costs[t * BB + b] = lse - row[yi];
    }
}

// Softmax of last-step logits into d_out[0 .. B*V).
__global__ void softmax_out(const float* __restrict__ logits, float* __restrict__ out)
{
    __shared__ float red[256];
    int b = blockIdx.x, tid = threadIdx.x;
    const float* row = logits + (size_t)b * VV;
    float mx = -1e30f;
    for (int v = tid; v < VV; v += 256) mx = fmaxf(mx, row[v]);
    red[tid] = mx; __syncthreads();
    for (int o = 128; o; o >>= 1) { if (tid < o) red[tid] = fmaxf(red[tid], red[tid + o]); __syncthreads(); }
    mx = red[0]; __syncthreads();
    float s = 0.0f;
    for (int v = tid; v < VV; v += 256) s += __expf(row[v] - mx);
    red[tid] = s; __syncthreads();
    for (int o = 128; o; o >>= 1) { if (tid < o) red[tid] += red[tid + o]; __syncthreads(); }
    float inv = 1.0f / red[0];
    for (int v = tid; v < VV; v += 256)
        out[(size_t)b * VV + v] = __expf(row[v] - mx) * inv;
}

// Final scalars: cost_per_batch and cost_coverage.  Single wave.
__global__ void finalize(const float* __restrict__ costs, const float* __restrict__ ymask,
                         const float* __restrict__ covacc, float* __restrict__ out)
{
    int lane = (int)(threadIdx.x & 31);
    float cb = 0.0f;
    if (lane < BB) {
        float num = 0.0f, den = 0.0f;
        for (int t = 0; t < TYY; ++t) {
            float ym = ymask[t * BB + lane];
            num += costs[t * BB + lane] * ym;
            den += ym;
        }
        cb = num / den;
    }
    for (int o = 8; o; o >>= 1) cb += __shfl_down(cb, o, 32);
    if (lane == 0) {
        out[(size_t)BB * VV]     = cb / (float)BB;
        out[(size_t)BB * VV + 1] = covacc[0] / (float)(TYY * BB);
    }
}

// ---------------------------------------------------------------------------
// Host driver.
// ---------------------------------------------------------------------------
extern "C" void kernel_launch(void* const* d_in, const int* in_sizes, int n_in,
                              void* d_out, int out_size, void* d_ws, size_t ws_size,
                              hipStream_t stream)
{
    (void)in_sizes; (void)n_in; (void)out_size; (void)ws_size;

    const float* y_emb   = (const float*)d_in[0];
    const float* context = (const float*)d_in[1];
    const float* h0      = (const float*)d_in[2];
    const float* c0      = (const float*)d_in[3];
    const float* x_mask  = (const float*)d_in[4];
    const float* y_mask  = (const float*)d_in[5];
    const float* cov0    = (const float*)d_in[6];
    const int*   y_idx   = (const int*)d_in[7];
    const float* W_ih    = (const float*)d_in[8];
    const float* W_hh    = (const float*)d_in[9];
    const float* b_ih    = (const float*)d_in[10];
    const float* b_hh    = (const float*)d_in[11];
    const float* Wx      = (const float*)d_in[12];
    const float* Ux      = (const float*)d_in[13];
    const float* bx      = (const float*)d_in[14];
    const float* Wc_att  = (const float*)d_in[15];
    const float* b_att   = (const float*)d_in[16];
    const float* W_comb  = (const float*)d_in[17];
    const float* U_att   = (const float*)d_in[18];
    const float* W_cov   = (const float*)d_in[19];
    const float* Wp      = (const float*)d_in[20];
    const float* bp      = (const float*)d_in[21];
    float* out = (float*)d_out;

    // Workspace carve-up (256B aligned).
    char* base = (char*)d_ws;
    size_t off = 0;
    auto carve = [&](size_t bytes) -> void* {
        void* p = base + off;
        off = (off + bytes + 255) & ~(size_t)255;
        return p;
    };
    bf16_t* wp_bf    = (bf16_t*)carve((size_t)VV * (HH + CC + ININ) * 2);   // 53.8 MB, L2-resident
    bf16_t* ctx_bf   = (bf16_t*)carve((size_t)TXX * BB * CC * 2);
    bf16_t* wc_bf    = (bf16_t*)carve((size_t)CC * CC * 2);
    bf16_t* wcomb_bf = (bf16_t*)carve((size_t)CC * (2 * HH) * 2);
    bf16_t* wg1_bf   = (bf16_t*)carve((size_t)4 * HH * (ININ + HH) * 2);    // [W_ih|W_hh]
    bf16_t* wg2_bf   = (bf16_t*)carve((size_t)4 * HH * (HH + CC) * 2);      // [Ux|Wx]
    bf16_t* yemb_bf  = (bf16_t*)carve((size_t)TYY * BB * ININ * 2);
    float*  pctx     = (float*)carve((size_t)TXX * BB * CC * 4);
    float*  logits   = (float*)carve((size_t)BB * VV * 4);
    float*  gates    = (float*)carve((size_t)BB * 4 * HH * 4);
    float*  ifoc     = (float*)carve((size_t)BB * 4 * HH * 4);
    float*  hq       = (float*)carve((size_t)BB * CC * 4);
    float*  sc       = (float*)carve((size_t)TXX * BB * 4);
    float*  h        = (float*)carve((size_t)BB * HH * 4);
    float*  c        = (float*)carve((size_t)BB * HH * 4);
    float*  h1       = (float*)carve((size_t)BB * HH * 4);
    float*  c1       = (float*)carve((size_t)BB * HH * 4);
    float*  acc      = (float*)carve((size_t)BB * TXX * 4);
    float*  costs    = (float*)carve((size_t)TYY * BB * 4);
    float*  blstm    = (float*)carve((size_t)4 * HH * 4);
    float*  covacc   = (float*)carve(256);
    bf16_t* xh       = (bf16_t*)carve((size_t)BB * (ININ + HH) * 2);
    bf16_t* s2       = (bf16_t*)carve((size_t)BB * (2 * HH) * 2);
    bf16_t* ha       = (bf16_t*)carve((size_t)BB * (HH + CC) * 2);
    bf16_t* feats    = (bf16_t*)carve((size_t)BB * (HH + CC + ININ) * 2);

    auto cblocks = [](size_t n) { return (unsigned)((n + 255) / 256); };
    // waves = tiles_m * (tiles_n / NT); 8 waves per 256-thread block
    auto gemm_grid = [](int tiles_m, int tiles_n, int nt) {
        return (unsigned)(((size_t)tiles_m * (tiles_n / nt) * 32 + 255) / 256);
    };
    dim3 blk(256);

    // ---- One-time conversions / packing ----
    cvt_bf16<<<cblocks((size_t)VV * (HH + CC + ININ)), blk, 0, stream>>>(Wp, wp_bf, VV * (HH + CC + ININ));
    cvt_bf16<<<cblocks((size_t)TXX * BB * CC), blk, 0, stream>>>(context, ctx_bf, TXX * BB * CC);
    cvt_bf16<<<cblocks((size_t)CC * CC), blk, 0, stream>>>(Wc_att, wc_bf, CC * CC);
    cvt_bf16<<<cblocks((size_t)CC * 2 * HH), blk, 0, stream>>>(W_comb, wcomb_bf, CC * 2 * HH);
    cvt_bf16<<<cblocks((size_t)TYY * BB * ININ), blk, 0, stream>>>(y_emb, yemb_bf, TYY * BB * ININ);
    pack2_bf16<<<cblocks((size_t)4 * HH * (ININ + HH)), blk, 0, stream>>>(W_ih, ININ, W_hh, HH, wg1_bf, 4 * HH);
    pack2_bf16<<<cblocks((size_t)4 * HH * (HH + CC)), blk, 0, stream>>>(Ux, HH, Wx, CC, wg2_bf, 4 * HH);
    init_state<<<16, blk, 0, stream>>>(h0, c0, cov0, b_ih, b_hh, h, c, acc, blstm, covacc);

    // pctx = context @ Wc_att.T + b_att   (2560 x 512, K=512), NT=4
    wmma_gemm_bias<4><<<gemm_grid(TXX * BB / 16, CC / 16, 4), blk, 0, stream>>>(
        ctx_bf, CC, wc_bf, CC, b_att, pctx, CC, TXX * BB, CC, CC);

    // ---- 40 sequential decode steps ----
    for (int t = 0; t < TYY; ++t) {
        prep_step<<<cblocks((size_t)BB * (ININ + HH)), blk, 0, stream>>>(yemb_bf, h, xh, feats, t);

        // gates = [y_t|h] @ [W_ih|W_hh].T + (b_ih+b_hh)   (16 x 1024, K=384), NT=4
        wmma_gemm_bias<4><<<gemm_grid(1, 4 * HH / 16, 4), blk, 0, stream>>>(
            xh, ININ + HH, wg1_bf, ININ + HH, blstm, gates, 4 * HH, BB, 4 * HH, ININ + HH);

        lstm_cell1<<<cblocks((size_t)BB * HH), blk, 0, stream>>>(
            gates, h, c, y_mask, h1, c1, s2, ha, t);

        // hq = [h1|c1] @ W_comb_att.T   (16 x 512, K=512), NT=4
        wmma_gemm_bias<4><<<gemm_grid(1, CC / 16, 4), blk, 0, stream>>>(
            s2, 2 * HH, wcomb_bf, 2 * HH, (const float*)nullptr, hq, CC, BB, CC, 2 * HH);

        attn_score<<<(TXX * BB) / 8, blk, 0, stream>>>(
            pctx, hq, acc, W_cov, U_att, x_mask, sc);

        attn_soft<<<BB, blk, 0, stream>>>(sc, x_mask, context, acc, ha, feats, covacc);

        // ifoc = [h1|atted] @ [Ux|Wx].T + bx   (16 x 1024, K=768), NT=4
        wmma_gemm_bias<4><<<gemm_grid(1, 4 * HH / 16, 4), blk, 0, stream>>>(
            ha, HH + CC, wg2_bf, HH + CC, bx, ifoc, 4 * HH, BB, 4 * HH, HH + CC);

        lstm_cell2<<<cblocks((size_t)BB * HH), blk, 0, stream>>>(
            ifoc, h1, c1, y_mask, h, c, feats, t);

        // logits = [h2|atted|y_t] @ Wp.T + bp   (16 x 30000, K=896) — dominant.
        // 1875 N-tiles = 3 * 625 -> NT=3, 625 waves x 84 WMMAs each.
        wmma_gemm_bias<3><<<gemm_grid(1, VV / 16, 3), blk, 0, stream>>>(
            feats, HH + CC + ININ, wp_bf, HH + CC + ININ, bp, logits, VV, BB, VV, HH + CC + ININ);

        nll_cost<<<BB, blk, 0, stream>>>(logits, y_idx, costs, t);
    }

    // ---- Outputs ----
    softmax_out<<<BB, blk, 0, stream>>>(logits, out);
    finalize<<<1, 32, 0, stream>>>(costs, y_mask, covacc, out);
}
